// MessagePassingLayer_30262339568006
// MI455X (gfx1250) — matrix operand order repaired and checked by reading, not compiled, not run on hardware
//
#include <hip/hip_runtime.h>
#include <hip/hip_fp16.h>

typedef _Float16 h2   __attribute__((ext_vector_type(2)));
typedef _Float16 v8h  __attribute__((ext_vector_type(8)));
typedef _Float16 v16h __attribute__((ext_vector_type(16)));
typedef float    v8f  __attribute__((ext_vector_type(8)));

#define N_NODES   10000
#define HDIM      128
#define FDIM      32
#define MDIM      256
#define KDIM      288                 // F + 2H
#define ROWS      64                  // edges per tile
#define ROW_BYTES 592                 // 288 f16 = 576B, padded to 592B (bank-conflict-free, 16B aligned)
#define W_LDS_BYTES (MDIM * ROW_BYTES)          // 151552: W^T, row n = output column n, K-contiguous
#define X_LDS_OFF   W_LDS_BYTES
#define X_LDS_BYTES (ROWS * ROW_BYTES)          // 37888
#define TGT_OFF     (X_LDS_OFF + X_LDS_BYTES)
#define SRC_OFF     (TGT_OFF + ROWS * 4)
#define LDS_TOTAL   (SRC_OFF + ROWS * 4)        // 189952 bytes < 320KB WGP LDS

__global__ void zero_f32_kernel(float* __restrict__ p, size_t n4) {
  size_t i = (size_t)blockIdx.x * blockDim.x + threadIdx.x;
  if (i < n4) ((float4*)p)[i] = make_float4(0.f, 0.f, 0.f, 0.f);
}

// A 16x32 f16 fragment: lanes 0-15 row M=lane: K0..15 contiguous;
//                       lanes 16-31 same rows: K16..31 (caller adds +32B)
__device__ __forceinline__ v16h ldsFragA(const char* p) {
  v8h lo = *(const v8h*)(p);
  v8h hi = *(const v8h*)(p + 16);
  v16h r;
  #pragma unroll
  for (int i = 0; i < 8; ++i) { r[i] = lo[i]; r[8 + i] = hi[i]; }
  return r;
}
// B 32x16 f16 fragment: lanes 0-15 col N=lane: K0..15 contiguous; lanes 16-31: K16..31 (+32B by caller)
__device__ __forceinline__ v16h ldsFragB(const char* p) {
  v8h lo = *(const v8h*)(p);
  v8h hi = *(const v8h*)(p + 16);
  v16h r;
  #pragma unroll
  for (int i = 0; i < 8; ++i) { r[i] = lo[i]; r[8 + i] = hi[i]; }
  return r;
}

__global__ void __launch_bounds__(256)
mp_wmma_kernel(const float* __restrict__ hid, const float* __restrict__ ef,
               const int* __restrict__ srcI, const int* __restrict__ tgtI,
               const float* __restrict__ Wg, const float* __restrict__ bg,
               float* __restrict__ out, int B, int E, int totalTiles, int tilesPerBatch)
{
  extern __shared__ char smem[];
  char* Xlds   = smem + X_LDS_OFF;
  int*  tgtLds = (int*)(smem + TGT_OFF);
  int*  srcLds = (int*)(smem + SRC_OFF);

  const int tid = threadIdx.x;

  // ---- Stage W^T into LDS as f16 (row n = column n of W, K contiguous) ----
  #pragma unroll 1
  for (int it = 0; it < (KDIM * (MDIM / 2)) / 256; ++it) {   // 144 iters
    int idx = it * 256 + tid;          // 0..36863
    int k   = idx >> 7;                // 0..287
    int n   = (idx & 127) * 2;         // 0..254
    float2 w = *(const float2*)(Wg + (size_t)k * MDIM + n);
    *(_Float16*)(smem + (size_t)n       * ROW_BYTES + k * 2) = (_Float16)w.x;
    *(_Float16*)(smem + (size_t)(n + 1) * ROW_BYTES + k * 2) = (_Float16)w.y;
  }

  const int lane    = tid & 31;
  const int wave    = tid >> 5;        // 0..7 (wave32)
  const int laneLo  = lane & 15;
  const int laneHi  = lane >> 4;       // 0 or 1
  // per-wave output tile: 32 rows (2 row-groups) x 64 cols (4 col-tiles) -> best LDS reuse
  const int rowPair = wave & 1;        // rows rowPair*32 .. +31
  const int colQuad = wave >> 1;       // cols colQuad*64 .. +63

  // ---- tile-invariant values ----
  const char* aBase = Xlds + (size_t)(rowPair * 32 + laneLo) * ROW_BYTES + laneHi * 32;
  const char* bBase = smem + (size_t)(colQuad * 64 + laneLo) * ROW_BYTES + laneHi * 32;

  float bias[4];
  int   colIdx[4];
  #pragma unroll
  for (int t = 0; t < 4; ++t) {
    colIdx[t] = (colQuad * 4 + t) * 16 + laneLo;
    bias[t]   = bg[colIdx[t]];
  }
  const int rBase = rowPair * 32 + laneHi * 8;   // rows rBase..+7 and rBase+16..+23

  #pragma unroll 1
  for (int tile = blockIdx.x; tile < totalTiles; tile += gridDim.x) {
    __syncthreads();                    // protect LDS vs previous iteration readers
    const int b     = tile / tilesPerBatch;
    const int eBase = (tile - b * tilesPerBatch) * ROWS;
    const bool fullTile = (eBase + ROWS) <= E;

    // ---- stage edge indices ----
    if (tid < ROWS) {
      int e = eBase + tid;
      int s = 0, t = 0;
      if (e < E) { s = srcI[(size_t)b * E + e]; t = tgtI[(size_t)b * E + e]; }
      srcLds[tid] = s;
      tgtLds[tid] = t;
    }
    __syncthreads();

    // ---- gather + f32->f16 pack into X tile [64 x 288]; three branch-free passes ----
    {
      const float* efB  = ef  + ((size_t)b * E + eBase) * FDIM;
      const float* hidB = hid + (size_t)b * N_NODES * HDIM;
      // pass 1: edge features, cols [0,32): 64 rows x 16 pairs
      #pragma unroll 1
      for (int it = 0; it < 4; ++it) {
        int idx  = it * 256 + tid;
        int row  = idx >> 4;
        int col  = (idx & 15) * 2;
        float2 v = (fullTile || (eBase + row) < E)
                     ? *(const float2*)(efB + (size_t)row * FDIM + col)
                     : make_float2(0.f, 0.f);
        h2 p; p.x = (_Float16)v.x; p.y = (_Float16)v.y;
        *(h2*)(Xlds + (size_t)row * ROW_BYTES + col * 2) = p;
      }
      // pass 2: h[src], cols [32,160): 64 rows x 64 pairs
      #pragma unroll 1
      for (int it = 0; it < 16; ++it) {
        int idx  = it * 256 + tid;
        int row  = idx >> 6;
        int col  = (idx & 63) * 2;
        float2 v = (fullTile || (eBase + row) < E)
                     ? *(const float2*)(hidB + (size_t)srcLds[row] * HDIM + col)
                     : make_float2(0.f, 0.f);
        h2 p; p.x = (_Float16)v.x; p.y = (_Float16)v.y;
        *(h2*)(Xlds + (size_t)row * ROW_BYTES + (FDIM + col) * 2) = p;
      }
      // pass 3: h[tgt], cols [160,288)
      #pragma unroll 1
      for (int it = 0; it < 16; ++it) {
        int idx  = it * 256 + tid;
        int row  = idx >> 6;
        int col  = (idx & 63) * 2;
        float2 v = (fullTile || (eBase + row) < E)
                     ? *(const float2*)(hidB + (size_t)tgtLds[row] * HDIM + col)
                     : make_float2(0.f, 0.f);
        h2 p; p.x = (_Float16)v.x; p.y = (_Float16)v.y;
        *(h2*)(Xlds + (size_t)row * ROW_BYTES + (FDIM + HDIM + col) * 2) = p;
      }
    }
    __syncthreads();

    // ---- prefetch next tile's streaming data so its gather hits cache (overlaps WMMA) ----
    {
      int nextTile = tile + gridDim.x;
      if (nextTile < totalTiles) {
        int b2  = nextTile / tilesPerBatch;
        int eb2 = (nextTile - b2 * tilesPerBatch) * ROWS;
        // one 128B edge-feature row per thread (exactly one cache line)
        if (tid < ROWS && (eb2 + tid) < E) {
          const char* p = (const char*)(ef + ((size_t)b2 * E + eb2 + tid) * FDIM);
          __builtin_prefetch(p, 0, 3);
        }
        // index arrays: 64 ints each = 256B
        if (tid >= ROWS && tid < ROWS + 4) {
          __builtin_prefetch((const char*)(srcI + (size_t)b2 * E + eb2) + (tid - ROWS) * 64, 0, 3);
          __builtin_prefetch((const char*)(tgtI + (size_t)b2 * E + eb2) + (tid - ROWS) * 64, 0, 3);
        }
      }
    }

    // ---- WMMA: each wave computes 2x4 tiles of 16x16 (A reused x4, B reused x2) ----
    v8f acc[2][4];
    #pragma unroll
    for (int g = 0; g < 2; ++g)
      #pragma unroll
      for (int t = 0; t < 4; ++t)
        #pragma unroll
        for (int i = 0; i < 8; ++i) acc[g][t][i] = 0.0f;

    #pragma unroll 1
    for (int kc = 0; kc < KDIM / 32; ++kc) {     // 9 K-chunks; 12 ds_load_b128 per 8 WMMA
      v16h a0 = ldsFragA(aBase + kc * 64);
      v16h a1 = ldsFragA(aBase + 16 * ROW_BYTES + kc * 64);
      v16h bb[4];
      #pragma unroll
      for (int t = 0; t < 4; ++t)
        bb[t] = ldsFragB(bBase + t * 16 * ROW_BYTES + kc * 64);
      #pragma unroll
      for (int t = 0; t < 4; ++t) {
        acc[0][t] = __builtin_amdgcn_wmma_f32_16x16x32_f16(
            false, a0, false, bb[t], (short)0, acc[0][t], false, false);
        acc[1][t] = __builtin_amdgcn_wmma_f32_16x16x32_f16(
            false, a1, false, bb[t], (short)0, acc[1][t], false, false);
      }
    }

    // ---- epilogue: bias + ReLU + scatter-add by target node ----
    // C/D layout: VGPR i, lanes 0-15 -> M=i, N=lane; lanes 16-31 -> M=i+8, N=lane-16
    float* outB = out + (size_t)b * N_NODES * MDIM;
    int nodeOff[2][8];
    #pragma unroll
    for (int i = 0; i < 8; ++i) {
      nodeOff[0][i] = tgtLds[rBase + i]      * MDIM;   // < 2.56M, fits int
      nodeOff[1][i] = tgtLds[rBase + 16 + i] * MDIM;
    }

    if (fullTile) {
      // uniform path: no per-element EXEC manipulation
      #pragma unroll
      for (int g = 0; g < 2; ++g) {
        #pragma unroll
        for (int t = 0; t < 4; ++t) {
          #pragma unroll
          for (int i = 0; i < 8; ++i) {
            float v = acc[g][t][i] + bias[t];
            v = v > 0.0f ? v : 0.0f;
            unsafeAtomicAdd(outB + nodeOff[g][i] + colIdx[t], v);
          }
        }
      }
    } else {
      #pragma unroll
      for (int g = 0; g < 2; ++g) {
        #pragma unroll
        for (int t = 0; t < 4; ++t) {
          #pragma unroll
          for (int i = 0; i < 8; ++i) {
            if (eBase + rBase + g * 16 + i < E) {
              float v = acc[g][t][i] + bias[t];
              v = v > 0.0f ? v : 0.0f;
              unsafeAtomicAdd(outB + nodeOff[g][i] + colIdx[t], v);
            }
          }
        }
      }
    }
  }
}

extern "C" void kernel_launch(void* const* d_in, const int* in_sizes, int n_in,
                              void* d_out, int out_size, void* d_ws, size_t ws_size,
                              hipStream_t stream) {
  const float* hid  = (const float*)d_in[0];   // [B, N, H] f32
  const float* ef   = (const float*)d_in[1];   // [B, E, F] f32
  const int*   srcI = (const int*)d_in[2];     // [B, E]
  const int*   tgtI = (const int*)d_in[3];     // [B, E]
  const float* Wg   = (const float*)d_in[4];   // [288, 256] f32
  const float* bg   = (const float*)d_in[5];   // [256] f32
  float* out = (float*)d_out;                  // [B, N, 256] f32

  int B = in_sizes[0] / (N_NODES * HDIM);
  int E = in_sizes[2] / B;
  int tilesPerBatch = (E + ROWS - 1) / ROWS;
  int totalTiles = B * tilesPerBatch;

  // zero output (harness poisons it; we accumulate with atomics). out_size = B*N*256, /4 exact.
  size_t n4 = (size_t)out_size / 4;
  zero_f32_kernel<<<(unsigned)((n4 + 255) / 256), 256, 0, stream>>>(out, n4);

  (void)hipFuncSetAttribute((const void*)mp_wmma_kernel,
                            hipFuncAttributeMaxDynamicSharedMemorySize, LDS_TOTAL);

  int grid = totalTiles < 512 ? totalTiles : 512;
  if (grid < 1) grid = 1;
  mp_wmma_kernel<<<grid, 256, LDS_TOTAL, stream>>>(hid, ef, srcI, tgtI, Wg, bg, out,
                                                   B, E, totalTiles, tilesPerBatch);
}
